// MultiProtoAsWCosWithCrossAttn_15479062135505
// MI455X (gfx1250) — matrix operand order repaired and checked by reading, not compiled, not run on hardware
//
#include <hip/hip_runtime.h>
#include <hip/hip_bf16.h>

typedef __attribute__((ext_vector_type(16))) _Float16 v16h;
typedef __attribute__((ext_vector_type(8)))  _Float16 v8h;
typedef __attribute__((ext_vector_type(8)))  float    v8f;

#define EPS   1e-4f
#define TAU   20.0f
#define NPIX  16384      // 128*128
#define NCH   256
#define NPROT 320        // 5 shots * 64 grid cells
#define MTILES 1024      // NPIX/16
#define NTILES 20        // NPROT/16
#define KSTEPS 8         // NCH/32

// ---------------------------------------------------------------------------
// Kernel 1: avg-pool sup_x (5,256,128,128) over 16x16 blocks.
// One wave per output element (s,c,g); writes pooled[(s*64+g)*256 + c].
// ---------------------------------------------------------------------------
__global__ __launch_bounds__(256) void
k_avgpool(const float* __restrict__ sup, float* __restrict__ pooled) {
    int wid  = blockIdx.x * (blockDim.x >> 5) + (threadIdx.x >> 5);   // 0..81919
    int lane = threadIdx.x & 31;
    int s   = wid / (NCH * 64);
    int rem = wid - s * (NCH * 64);
    int c   = rem >> 6;
    int g   = rem & 63;
    int gy  = g >> 3, gx = g & 7;
    int base = ((s * NCH + c) * 128 + gy * 16) * 128 + gx * 16;
    int w0 = lane & 15;
    int h0 = lane >> 4;              // 0 or 1
    float sum = 0.f;
#pragma unroll
    for (int i = 0; i < 8; ++i) {
        int h = h0 + 2 * i;
        sum += sup[base + h * 128 + w0];
    }
#pragma unroll
    for (int off = 16; off >= 1; off >>= 1)
        sum += __shfl_xor(sum, off, 32);
    if (lane == 0)
        pooled[(s * 64 + g) * NCH + c] = sum * (1.0f / 256.0f);
}

// ---------------------------------------------------------------------------
// Kernel 2: mean-center + normalize each proto row, emit f16 B-fragments.
// B frag layout (contiguous k, as in A; 16-bit B 32x16): B_ws[t][s][lane][i] holds
// pro_n[p = t*16 + (lane&15)][k = s*32 + ((lane>>4)*16) + i],  i=0..15.
// One wave per proto row p; lane holds channels [8*lane, 8*lane+8).
// ---------------------------------------------------------------------------
__global__ __launch_bounds__(256) void
k_protos(const float* __restrict__ pooled, _Float16* __restrict__ Bws) {
    int p    = blockIdx.x * (blockDim.x >> 5) + (threadIdx.x >> 5);   // 0..319
    int lane = threadIdx.x & 31;
    float v[8];
    int base = p * NCH + lane * 8;
    float sum = 0.f;
#pragma unroll
    for (int j = 0; j < 8; ++j) { v[j] = pooled[base + j]; sum += v[j]; }
#pragma unroll
    for (int off = 16; off >= 1; off >>= 1)
        sum += __shfl_xor(sum, off, 32);
    float mean = sum * (1.0f / 256.0f);
    float ss = 0.f;
#pragma unroll
    for (int j = 0; j < 8; ++j) { v[j] -= mean; ss += v[j] * v[j]; }
#pragma unroll
    for (int off = 16; off >= 1; off >>= 1)
        ss += __shfl_xor(ss, off, 32);
    float inv = 1.0f / fmaxf(sqrtf(ss), EPS);
    v8h h;
#pragma unroll
    for (int j = 0; j < 8; ++j) h[j] = (_Float16)(v[j] * inv);
    int t   = p >> 4, row = p & 15;
    int s5  = lane >> 2;             // k-step 0..7
    int ko  = (lane & 3) * 8;        // 0,8,16,24 within k-step
    int dl  = row + (ko & 16);       // dest lane in fragment
    int pos = ko & 15;               // element position 0 or 8
    *(v8h*)(Bws + (((t * 8 + s5) * 32 + dl) * 16 + pos)) = h;
}

// ---------------------------------------------------------------------------
// Kernel 3: mean-center + normalize query per pixel over channels, emit f16
// A-fragments. A frag layout (16-bit A 16x32): lane l<16 holds row l,
// K=0..15; lane l>=16 holds row l-16, K=16..31 (contiguous k).
// One thread per pixel; coalesced channel reads; 16B fragment stores.
// ---------------------------------------------------------------------------
__global__ __launch_bounds__(256) void
k_qnorm(const float* __restrict__ qry, _Float16* __restrict__ Aws) {
    int p = blockIdx.x * blockDim.x + threadIdx.x;    // pixel 0..16383
    float sum = 0.f, sq = 0.f;
    for (int c = 0; c < NCH; ++c) {
        float v = qry[c * NPIX + p];
        sum += v; sq += v * v;
    }
    float mean = sum * (1.0f / 256.0f);
    float n2   = fmaxf(sq - sum * mean, 0.0f);        // ||x - mean||^2
    float inv  = 1.0f / fmaxf(sqrtf(n2), EPS);
    int tile = p >> 4, row = p & 15;
#pragma unroll
    for (int s = 0; s < KSTEPS; ++s) {
#pragma unroll
        for (int grp = 0; grp < 4; ++grp) {
            int kb = s * 32 + grp * 8;
            v8h h;
#pragma unroll
            for (int j = 0; j < 8; ++j)
                h[j] = (_Float16)((qry[(kb + j) * NPIX + p] - mean) * inv);
            int laneD = row + ((grp >> 1) << 4);
            int pos   = (grp & 1) * 8;
            *(v8h*)(Aws + (((tile * 8 + s) * 32 + laneD) * 16 + pos)) = h;
        }
    }
}

// ---------------------------------------------------------------------------
// Kernel 4: WMMA GEMM [16 pix x 256] x [256 x 320] + fused TAU * softmax
// expectation (== pred_grid) and first-occurrence argmax (== debug_assign).
// One wave per 16-pixel M-tile; 8 waves/block share the per-k-step B slice
// through LDS (20 KB), cutting global B traffic 8x. __launch_bounds__(256,1)
// grants the full VGPR budget so the 20 accumulator tiles stay register-
// resident (no scratch spills).
// ---------------------------------------------------------------------------
__global__ __launch_bounds__(256, 1) void
k_gemm_softmax(const _Float16* __restrict__ Aws,
               const _Float16* __restrict__ Bws,
               float* __restrict__ outPred,
               float* __restrict__ outAssign) {
    __shared__ _Float16 Bsh[NTILES * 32 * 16];        // 20 KB: one k-step slice

    int tid  = threadIdx.x;
    int tile = blockIdx.x * (blockDim.x >> 5) + (tid >> 5);           // 0..1023
    int lane = tid & 31;

    const v16h* Af = (const v16h*)(Aws) + (size_t)tile * (KSTEPS * 32);
    const v8h*  Bg = (const v8h*)(Bws);

    v8f acc[NTILES];
#pragma unroll
    for (int t = 0; t < NTILES; ++t) acc[t] = (v8f)(0.0f);

    for (int s = 0; s < KSTEPS; ++s) {
        // Cooperative stage of the k-step-s B slice: 1280 x 16B chunks.
#pragma unroll
        for (int i = 0; i < 5; ++i) {
            int cid = tid + i * 256;                  // 0..1279
            int t   = cid >> 6;
            int rem = cid & 63;
            int ln  = rem >> 1;
            int hf  = rem & 1;
            ((v8h*)Bsh)[(t * 32 + ln) * 2 + hf] =
                Bg[((t * 8 + s) * 32 + ln) * 2 + hf];
        }
        __syncthreads();
        v16h a = Af[s * 32 + lane];
#pragma unroll
        for (int t = 0; t < NTILES; ++t) {
            v16h b = ((const v16h*)Bsh)[t * 32 + lane];
            acc[t] = __builtin_amdgcn_wmma_f32_16x16x32_f16(
                false, a, false, b, (short)0, acc[t], false, false);
        }
        __syncthreads();
    }

    // C layout: lane holds col n=(lane&15); VGPR r holds row m=r+8*(lane>>4).
    int col    = lane & 15;
    int halfId = lane >> 4;
#pragma unroll
    for (int r = 0; r < 8; ++r) {
        float m = -3.4e38f;
        int   bi = 0;
#pragma unroll
        for (int t = 0; t < NTILES; ++t) {
            float d = acc[t][r] * TAU;
            if (d > m) { m = d; bi = t * 16 + col; }
        }
        // max + argmax across the 16 lanes of this half (smaller index wins ties)
#pragma unroll
        for (int off = 1; off <= 8; off <<= 1) {
            float om = __shfl_xor(m, off, 32);
            int   ob = __shfl_xor(bi, off, 32);
            if (om > m || (om == m && ob < bi)) { m = om; bi = ob; }
        }
        float se = 0.f, sed = 0.f;
#pragma unroll
        for (int t = 0; t < NTILES; ++t) {
            float d = acc[t][r] * TAU;
            float e = __expf(d - m);
            se  += e;
            sed += e * d;
        }
#pragma unroll
        for (int off = 1; off <= 8; off <<= 1) {
            se  += __shfl_xor(se,  off, 32);
            sed += __shfl_xor(sed, off, 32);
        }
        if (col == 0) {
            int pixel = tile * 16 + halfId * 8 + r;
            outPred[pixel]   = sed / se;       // sum_p softmax(d)_p * d_p
            outAssign[pixel] = (float)bi;      // argmax_p d_p
        }
    }
}

// ---------------------------------------------------------------------------
extern "C" void kernel_launch(void* const* d_in, const int* in_sizes, int n_in,
                              void* d_out, int out_size, void* d_ws, size_t ws_size,
                              hipStream_t stream) {
    const float* qry  = (const float*)d_in[0];   // (1,1,256,128,128)
    const float* supx = (const float*)d_in[1];   // (1,5,1,256,128,128)
    // d_in[2] (sup_y) and d_in[3] (thresh) are unused by the reference math.

    float*    pooled = (float*)d_ws;                                  // 320*256 f32
    _Float16* Bws    = (_Float16*)((char*)d_ws + (512u << 10));       // 320*256 f16 frags
    _Float16* Aws    = (_Float16*)((char*)d_ws + (1024u << 10));      // 16384*256 f16 frags

    float* outPred   = (float*)d_out;            // pred_grid  (1,1,128,128)
    float* outAssign = outPred + NPIX;           // debug_assign (1,128,128)

    // 81920 wave-outputs, 8 waves per 256-thread block
    k_avgpool<<<81920 / 8, 256, 0, stream>>>(supx, pooled);
    // 320 proto waves
    k_protos<<<NPROT / 8, 256, 0, stream>>>(pooled, Bws);
    // 16384 pixels, thread-per-pixel
    k_qnorm<<<NPIX / 256, 256, 0, stream>>>(qry, Aws);
    // 1024 M-tiles, 8 waves per block
    k_gemm_softmax<<<MTILES / 8, 256, 0, stream>>>(Aws, Bws, outPred, outAssign);
}
